// AttnCodebook_62594853372332
// MI455X (gfx1250) — compile-verified, matrix operand order checked by reading, not run on hardware
//
#include <hip/hip_runtime.h>
#include <cmath>

// ---------------------------------------------------------------------------
// Shapes (compile-time, from the reference)
// ---------------------------------------------------------------------------
#define BB 8
#define NN 2048
#define DD 768
#define CC 2048
#define HH 3072

// Padded LDS row stride (bf16 elements): 40 elems = 80 B, multiple of 16 B,
// and 20-bank stride -> 16 consecutive rows hit 16 distinct banks.
#define ASTRIDE 40

typedef __attribute__((ext_vector_type(16))) __bf16 v16bf;
typedef __attribute__((ext_vector_type(8)))  float  v8f;

// ---------------------------------------------------------------------------
// Helpers
// ---------------------------------------------------------------------------
__device__ __forceinline__ __bf16 to_bf16(float f) {
    union { float f; unsigned u; } in; in.f = f;
    unsigned r = in.u + 0x7FFFu + ((in.u >> 16) & 1u);   // round-to-nearest-even
    union { unsigned short s; __bf16 b; } out;
    out.s = (unsigned short)(r >> 16);
    return out.b;
}

__device__ __forceinline__ float hash_uniform(unsigned x) {
    x ^= x >> 17; x *= 0xED5AD4BBu;
    x ^= x >> 11; x *= 0xAC4C1B51u;
    x ^= x >> 15; x *= 0x31848BABu;
    x ^= x >> 14;
    return (float)(x >> 8) * (1.0f / 16777216.0f);
}

__device__ __forceinline__ void wait_async0() {
#if __has_builtin(__builtin_amdgcn_s_wait_asynccnt)
    __builtin_amdgcn_s_wait_asynccnt(0);
#else
    asm volatile("s_wait_asynccnt 0x0" ::: "memory");
#endif
}

// Async-copy a rows x 32 bf16 tile (row-major, leading dim ld) from global
// into LDS at byte offset lds_byte_base with padded row stride ASTRIDE.
// Each 16-byte chunk is one GLOBAL_LOAD_ASYNC_TO_LDS_B128 issue (ASYNCcnt).
__device__ __forceinline__ void async_tile(const __bf16* __restrict__ g, int ld,
                                           int row0, int k0, unsigned lds_byte_base,
                                           int rows, int tid)
{
    int nchunks = rows * 4;                       // 4 x 16B chunks per 32-elem row
    for (int ch = tid; ch < nchunks; ch += 256) {
        int r  = ch >> 2;
        int c8 = (ch & 3) << 3;                   // element offset within row
        unsigned laddr = lds_byte_base + (unsigned)(r * ASTRIDE + c8) * 2u;
        unsigned long long ga =
            (unsigned long long)(size_t)(g + (size_t)(row0 + r) * ld + k0 + c8);
        asm volatile("global_load_async_to_lds_b128 %0, %1, off"
                     :: "v"(laddr), "v"(ga)
                     : "memory");
    }
}

// Read one 16x32 bf16 fragment from an LDS tile (A-operand layout per CDNA5
// ISA 7.12.2): lanes 0-15 hold row m, K={0..7,16..23}; lanes 16-31 hold row m,
// K={8..15,24..31}. Two ds_load_b128 per lane. B operand is the
// transpose-symmetric layout, so reading the staged Bt tile with the same
// pattern produces the B fragment.
__device__ __forceinline__ v16bf lds_frag(const __bf16* sbase, int r0, int lane) {
    int m  = r0 + (lane & 15);
    int kh = ((lane >> 4) & 1) << 3;              // 0 or 8
    const __bf16* p = sbase + m * ASTRIDE + kh;
    v16bf f;
    ((float4*)&f)[0] = *(const float4*)(p);
    ((float4*)&f)[1] = *(const float4*)(p + 16);
    return f;
}

// ---------------------------------------------------------------------------
// Generic bf16 WMMA GEMM:  Out[M,Ncols] = epilogue( scale * A[M,K] x Bt[Ncols,K]^T )
// A row-major (lda), Bt row-major (ldb) == B^T. 256 threads = 8 waves, wave
// tile 32x32 (2x2 of 16x16), block tile 128x64. M%128==0, Ncols%64==0, K%32==0.
// K tiles are double-buffered in LDS via GLOBAL_LOAD_ASYNC_TO_LDS_B128 so the
// DMA of step k+1 overlaps the WMMAs of step k.
// ---------------------------------------------------------------------------
__global__ void __launch_bounds__(256)
gemm_bf16_wmma(const __bf16* __restrict__ A,  int lda,
               const __bf16* __restrict__ Bt, int ldb,
               int K, int ldo,
               float* __restrict__ outF, __bf16* __restrict__ outBf,
               float scale,
               const float* __restrict__ bias,        // per-column, may be null
               const float* __restrict__ resid,       // same layout as out, may be null
               const float* __restrict__ res_scale_p, // scalar ptr, null -> 1.0
               const float* __restrict__ silu_src)    // gate input, may be null
{
    // LDS: 2 stages of (A: 128 x ASTRIDE, B: 64 x ASTRIDE) bf16 = 30 KB
    __shared__ __bf16 smem[2 * (128 + 64) * ASTRIDE];
    const int AEL   = 128 * ASTRIDE;              // elements
    const int STAGE = (128 + 64) * ASTRIDE;       // elements per stage

    int tid  = threadIdx.x;
    int lane = tid & 31;
    int wave = tid >> 5;                          // 0..7
    int wrow = (wave >> 1) * 32;                  // wave tile inside block
    int wcol = (wave & 1) * 32;
    int brow = blockIdx.y * 128;
    int bcol = blockIdx.x * 64;

    unsigned sbase = (unsigned)(size_t)&smem[0];  // LDS byte address (addr[31:0])

    v8f acc00 = {0.f,0.f,0.f,0.f,0.f,0.f,0.f,0.f};
    v8f acc01 = acc00, acc10 = acc00, acc11 = acc00;

    // Prologue: stage 0
    async_tile(A,  lda, brow, 0, sbase,            128, tid);
    async_tile(Bt, ldb, bcol, 0, sbase + AEL * 2u, 64,  tid);
    wait_async0();
    __syncthreads();

    for (int k0 = 0; k0 < K; k0 += 32) {
        int cur = (k0 >> 5) & 1;
        int nxt = cur ^ 1;
        if (k0 + 32 < K) {  // kick off next stage DMA (overlaps WMMAs below)
            unsigned nb = sbase + (unsigned)(nxt * STAGE) * 2u;
            async_tile(A,  lda, brow, k0 + 32, nb,            128, tid);
            async_tile(Bt, ldb, bcol, k0 + 32, nb + AEL * 2u, 64,  tid);
        }
        const __bf16* sA = smem + cur * STAGE;
        const __bf16* sB = smem + cur * STAGE + AEL;

        v16bf fa0 = lds_frag(sA, wrow,      lane);
        v16bf fa1 = lds_frag(sA, wrow + 16, lane);
        v16bf fb0 = lds_frag(sB, wcol,      lane);
        v16bf fb1 = lds_frag(sB, wcol + 16, lane);
        acc00 = __builtin_amdgcn_wmma_f32_16x16x32_bf16(false, fa0, false, fb0, (short)0, acc00, false, false);
        acc01 = __builtin_amdgcn_wmma_f32_16x16x32_bf16(false, fa0, false, fb1, (short)0, acc01, false, false);
        acc10 = __builtin_amdgcn_wmma_f32_16x16x32_bf16(false, fa1, false, fb0, (short)0, acc10, false, false);
        acc11 = __builtin_amdgcn_wmma_f32_16x16x32_bf16(false, fa1, false, fb1, (short)0, acc11, false, false);

        if (k0 + 32 < K) wait_async0();           // own wave's DMA done
        __syncthreads();                          // everyone's DMA/read done
    }

    float rs = res_scale_p ? *res_scale_p : 1.0f;
    v8f accs[2][2] = {{acc00, acc01}, {acc10, acc11}};
    int row0 = brow + wrow;
    int col0 = bcol + wcol;

    #pragma unroll
    for (int ti = 0; ti < 2; ++ti)
    #pragma unroll
    for (int tj = 0; tj < 2; ++tj) {
        int col   = col0 + tj * 16 + (lane & 15);
        int rbase = row0 + ti * 16 + (((lane >> 4) & 1) << 3);
        v8f acc   = accs[ti][tj];
        #pragma unroll
        for (int e = 0; e < 8; ++e) {
            size_t idx = (size_t)(rbase + e) * ldo + col;
            float val = acc[e] * scale;
            if (bias)     val += bias[col];
            if (silu_src) { float g = silu_src[idx]; val *= g / (1.0f + __expf(-g)); }
            if (resid)    val += rs * resid[idx];
            if (outF)  outF[idx]  = val;
            if (outBf) outBf[idx] = to_bf16(val);
        }
    }
}

// ---------------------------------------------------------------------------
// RMSNorm (fp32 in, bf16 out), one block of 256 threads per row
// ---------------------------------------------------------------------------
__global__ void __launch_bounds__(256)
rmsnorm_bf16_kernel(const float* __restrict__ x, const float* __restrict__ scale,
                    __bf16* __restrict__ out, int cols)
{
    __shared__ float red[256];
    int row = blockIdx.x;
    const float* xr = x + (size_t)row * cols;
    float ss = 0.f;
    for (int i = threadIdx.x; i < cols; i += 256) { float v = xr[i]; ss += v * v; }
    red[threadIdx.x] = ss; __syncthreads();
    for (int s = 128; s > 0; s >>= 1) {
        if (threadIdx.x < s) red[threadIdx.x] += red[threadIdx.x + s];
        __syncthreads();
    }
    float inv = rsqrtf(red[0] / (float)cols + 1e-6f);
    for (int i = threadIdx.x; i < cols; i += 256)
        out[(size_t)row * cols + i] = to_bf16(xr[i] * inv * scale[i]);
}

// ---------------------------------------------------------------------------
// RoPE (fp32 in, bf16 out), one block per row; pos indexed by row
// ---------------------------------------------------------------------------
__global__ void __launch_bounds__(256)
rope_f32_to_bf16(const float* __restrict__ x, const int* __restrict__ pos,
                 __bf16* __restrict__ out, int half)
{
    int row = blockIdx.x;
    float p = (float)pos[row];
    const float* xr = x + (size_t)row * (2 * half);
    __bf16* o = out + (size_t)row * (2 * half);
    const float lnb = 9.210340371976184f;      // ln(10000)
    for (int i = threadIdx.x; i < half; i += 256) {
        float invf = __expf(-lnb * ((float)i / (float)half));
        float ang  = p * invf;
        float c, s; __sincosf(ang, &c, &s);
        float x1 = xr[i], x2 = xr[i + half];
        o[i]        = to_bf16(x1 * c - x2 * s);
        o[i + half] = to_bf16(x1 * s + x2 * c);
    }
}

// ---------------------------------------------------------------------------
// Gumbel + softmax over C=2048: one block of 256 threads per (b,n) row
// ---------------------------------------------------------------------------
__global__ void __launch_bounds__(256)
gumbel_softmax_kernel(const float* __restrict__ la, float* __restrict__ la_tau,
                      float* __restrict__ z, __bf16* __restrict__ z_bf,
                      const float* __restrict__ tau_p, const float* __restrict__ gns_p,
                      int batch)
{
    __shared__ float red[256];
    int n = blockIdx.x;
    const float* lar = la + (size_t)n * CC;
    float tau = *tau_p, gns = *gns_p;
    float vals[8];
    float mx = -1e30f;
    #pragma unroll
    for (int j = 0; j < 8; ++j) {
        int c = threadIdx.x + j * 256;
        unsigned h = ((unsigned)batch * 0x9E3779B1u) ^ ((unsigned)n * 0x85EBCA6Bu)
                   ^ ((unsigned)c * 0xC2B2AE35u);
        float u = hash_uniform(h);
        float g = -logf(-logf(u + 1e-10f) + 1e-10f);
        float v = (lar[c] + gns * g) / tau;
        vals[j] = v;
        mx = fmaxf(mx, v);
        la_tau[(size_t)n * CC + c] = v;
    }
    red[threadIdx.x] = mx; __syncthreads();
    for (int s = 128; s > 0; s >>= 1) {
        if (threadIdx.x < s) red[threadIdx.x] = fmaxf(red[threadIdx.x], red[threadIdx.x + s]);
        __syncthreads();
    }
    mx = red[0]; __syncthreads();
    float sum = 0.f;
    #pragma unroll
    for (int j = 0; j < 8; ++j) { vals[j] = __expf(vals[j] - mx); sum += vals[j]; }
    red[threadIdx.x] = sum; __syncthreads();
    for (int s = 128; s > 0; s >>= 1) {
        if (threadIdx.x < s) red[threadIdx.x] += red[threadIdx.x + s];
        __syncthreads();
    }
    float inv = 1.0f / red[0];
    #pragma unroll
    for (int j = 0; j < 8; ++j) {
        int c = threadIdx.x + j * 256;
        float zz = vals[j] * inv;
        z[(size_t)n * CC + c]    = zz;
        z_bf[(size_t)n * CC + c] = to_bf16(zz);
    }
}

// ---------------------------------------------------------------------------
// fp32 -> bf16 elementwise convert
// ---------------------------------------------------------------------------
__global__ void __launch_bounds__(256)
f32_to_bf16_kernel(const float* __restrict__ in, __bf16* __restrict__ out, int n)
{
    int i = blockIdx.x * 256 + threadIdx.x;
    if (i < n) out[i] = to_bf16(in[i]);
}

// ---------------------------------------------------------------------------
// Transposing convert: in fp32 (rows x cols) -> out bf16 (cols x rows)
// ---------------------------------------------------------------------------
__global__ void __launch_bounds__(256)
transpose_f32_to_bf16(const float* __restrict__ in, __bf16* __restrict__ out,
                      int rows, int cols)
{
    __shared__ float tile[32][33];
    int c0 = blockIdx.x * 32;
    int r0 = blockIdx.y * 32;
    for (int i = threadIdx.y; i < 32; i += 8)
        tile[i][threadIdx.x] = in[(size_t)(r0 + i) * cols + c0 + threadIdx.x];
    __syncthreads();
    for (int i = threadIdx.y; i < 32; i += 8)
        out[(size_t)(c0 + i) * rows + r0 + threadIdx.x] = to_bf16(tile[threadIdx.x][i]);
}

// ---------------------------------------------------------------------------
// Host launcher
// ---------------------------------------------------------------------------
static inline size_t alignup256(size_t x) { return (x + 255) & ~(size_t)255; }

extern "C" void kernel_launch(void* const* d_in, const int* in_sizes, int n_in,
                              void* d_out, int out_size, void* d_ws, size_t ws_size,
                              hipStream_t stream)
{
    (void)in_sizes; (void)n_in; (void)out_size; (void)ws_size;

    const float* latents    = (const float*)d_in[0];
    const float* tau_p      = (const float*)d_in[1];
    const float* res_sc_p   = (const float*)d_in[2];
    const float* gns_p      = (const float*)d_in[3];
    const int*   positions  = (const int*)  d_in[4];
    const float* codebook   = (const float*)d_in[5];
    const float* nctx_s     = (const float*)d_in[6];
    const float* nq_s       = (const float*)d_in[7];
    const float* nff_s      = (const float*)d_in[8];
    const float* Wq         = (const float*)d_in[9];
    const float* Wk         = (const float*)d_in[10];
    const float* Wv         = (const float*)d_in[11];
    const float* bv         = (const float*)d_in[12];
    const float* Wc         = (const float*)d_in[13];
    const float* W1         = (const float*)d_in[14];
    const float* W2         = (const float*)d_in[15];
    const float* W3         = (const float*)d_in[16];

    float* out    = (float*)d_out;
    float* out_cb = out;                                      // (B,N,D)
    float* out_la = out_cb + (size_t)BB * NN * DD;            // (B,N,C)
    float* out_lt = out_la + (size_t)BB * NN * CC;            // (B,N,C)
    float* out_z  = out_lt + (size_t)BB * NN * CC;            // (B,N,C)

    // ---- workspace carve-up -------------------------------------------------
    char* w = (char*)d_ws;
    size_t off = 0;
    auto alloc = [&](size_t bytes) -> void* {
        void* p = w + off; off = alignup256(off + bytes); return p;
    };
    __bf16* wq_bf   = (__bf16*)alloc((size_t)DD * DD * 2);
    __bf16* wk_bf   = (__bf16*)alloc((size_t)DD * DD * 2);
    __bf16* wv_bf   = (__bf16*)alloc((size_t)DD * DD * 2);
    __bf16* wc_bf   = (__bf16*)alloc((size_t)DD * DD * 2);
    __bf16* w1_bf   = (__bf16*)alloc((size_t)HH * DD * 2);
    __bf16* w3_bf   = (__bf16*)alloc((size_t)HH * DD * 2);
    __bf16* w2_bf   = (__bf16*)alloc((size_t)DD * HH * 2);
    __bf16* ctx_bf  = (__bf16*)alloc((size_t)CC * DD * 2);      // rmsnorm(codebook)
    __bf16* x_bf    = (__bf16*)alloc((size_t)BB * NN * DD * 2); // rmsnorm(latents)
    float*  k_f32   = (float*) alloc((size_t)CC * DD * 4);
    float*  v_f32   = (float*) alloc((size_t)CC * DD * 4);
    __bf16* vT_bf   = (__bf16*)alloc((size_t)DD * CC * 2);
    // per-batch reusable
    float*  q_f32   = (float*) alloc((size_t)NN * DD * 4);
    __bf16* q_bf    = (__bf16*)alloc((size_t)NN * DD * 2);
    __bf16* kr_bf   = (__bf16*)alloc((size_t)CC * DD * 2);
    __bf16* z_bf    = (__bf16*)alloc((size_t)NN * CC * 2);
    __bf16* attn_bf = (__bf16*)alloc((size_t)NN * DD * 2);
    float*  ao_f32  = (float*) alloc((size_t)NN * DD * 4);    // attn_output
    __bf16* h_bf    = (__bf16*)alloc((size_t)NN * DD * 2);
    float*  g1_f32  = (float*) alloc((size_t)NN * HH * 4);
    __bf16* gate_bf = (__bf16*)alloc((size_t)NN * HH * 2);

    const float inv_sqrt_d = 1.0f / sqrtf((float)DD);

    // ---- weight conversions -------------------------------------------------
    auto conv = [&](const float* src, __bf16* dst, int n) {
        f32_to_bf16_kernel<<<(n + 255) / 256, 256, 0, stream>>>(src, dst, n);
    };
    conv(Wq, wq_bf, DD * DD);
    conv(Wk, wk_bf, DD * DD);
    conv(Wv, wv_bf, DD * DD);
    conv(Wc, wc_bf, DD * DD);
    conv(W1, w1_bf, HH * DD);
    conv(W3, w3_bf, HH * DD);
    conv(W2, w2_bf, DD * HH);

    // ---- norms (context + queries) -----------------------------------------
    rmsnorm_bf16_kernel<<<CC, 256, 0, stream>>>(codebook, nctx_s, ctx_bf, DD);
    rmsnorm_bf16_kernel<<<BB * NN, 256, 0, stream>>>(latents, nq_s, x_bf, DD);

    // ---- k = ctx @ Wk^T ; v = ctx @ Wv^T + bv -------------------------------
    dim3 blk(256);
    {
        dim3 g(DD / 64, CC / 128);
        gemm_bf16_wmma<<<g, blk, 0, stream>>>(ctx_bf, DD, wk_bf, DD, DD, DD,
                                              k_f32, nullptr, 1.0f,
                                              nullptr, nullptr, nullptr, nullptr);
        gemm_bf16_wmma<<<g, blk, 0, stream>>>(ctx_bf, DD, wv_bf, DD, DD, DD,
                                              v_f32, nullptr, 1.0f,
                                              bv, nullptr, nullptr, nullptr);
    }
    // v^T (D x C) in bf16 for the z @ v GEMM
    {
        dim3 g(DD / 32, CC / 32);
        transpose_f32_to_bf16<<<g, dim3(32, 8), 0, stream>>>(v_f32, vT_bf, CC, DD);
    }

    // ---- per-batch pipeline -------------------------------------------------
    for (int b = 0; b < BB; ++b) {
        const __bf16* xb   = x_bf + (size_t)b * NN * DD;
        const float*  latb = latents + (size_t)b * NN * DD;
        const int*    posb = positions + (size_t)b * 2048;
        float* la_b = out_la + (size_t)b * NN * CC;
        float* lt_b = out_lt + (size_t)b * NN * CC;
        float* z_b  = out_z  + (size_t)b * NN * CC;
        float* cb_b = out_cb + (size_t)b * NN * DD;

        // q = normed_queries @ Wq^T  (fp32 for RoPE)
        {
            dim3 g(DD / 64, NN / 128);
            gemm_bf16_wmma<<<g, blk, 0, stream>>>(xb, DD, wq_bf, DD, DD, DD,
                                                  q_f32, nullptr, 1.0f,
                                                  nullptr, nullptr, nullptr, nullptr);
        }
        // RoPE(q), RoPE(k) -> bf16
        rope_f32_to_bf16<<<NN, 256, 0, stream>>>(q_f32, posb, q_bf, DD / 2);
        rope_f32_to_bf16<<<CC, 256, 0, stream>>>(k_f32, posb, kr_bf, DD / 2);

        // log_alpha = q @ k^T / sqrt(D)
        {
            dim3 g(CC / 64, NN / 128);
            gemm_bf16_wmma<<<g, blk, 0, stream>>>(q_bf, DD, kr_bf, DD, DD, CC,
                                                  la_b, nullptr, inv_sqrt_d,
                                                  nullptr, nullptr, nullptr, nullptr);
        }
        // gumbel noise + temperature + softmax
        gumbel_softmax_kernel<<<NN, 256, 0, stream>>>(la_b, lt_b, z_b, z_bf,
                                                      tau_p, gns_p, b);
        // attn = z @ v   (B-operand = v^T, row-major (D,C))
        {
            dim3 g(DD / 64, NN / 128);
            gemm_bf16_wmma<<<g, blk, 0, stream>>>(z_bf, CC, vT_bf, CC, CC, DD,
                                                  nullptr, attn_bf, 1.0f,
                                                  nullptr, nullptr, nullptr, nullptr);
        }
        // attn_output = residual_scaling * latents + attn @ Wc^T
        {
            dim3 g(DD / 64, NN / 128);
            gemm_bf16_wmma<<<g, blk, 0, stream>>>(attn_bf, DD, wc_bf, DD, DD, DD,
                                                  ao_f32, nullptr, 1.0f,
                                                  nullptr, latb, res_sc_p, nullptr);
        }
        // h = rmsnorm(attn_output)
        rmsnorm_bf16_kernel<<<NN, 256, 0, stream>>>(ao_f32, nff_s, h_bf, DD);

        // g1 = h @ W1^T ; gate = silu(g1) * (h @ W3^T) ; ff = gate @ W2^T + attn_output
        {
            dim3 g(HH / 64, NN / 128);
            gemm_bf16_wmma<<<g, blk, 0, stream>>>(h_bf, DD, w1_bf, DD, DD, HH,
                                                  g1_f32, nullptr, 1.0f,
                                                  nullptr, nullptr, nullptr, nullptr);
            gemm_bf16_wmma<<<g, blk, 0, stream>>>(h_bf, DD, w3_bf, DD, DD, HH,
                                                  nullptr, gate_bf, 1.0f,
                                                  nullptr, nullptr, nullptr, g1_f32);
        }
        {
            dim3 g(DD / 64, NN / 128);
            gemm_bf16_wmma<<<g, blk, 0, stream>>>(gate_bf, HH, w2_bf, HH, HH, DD,
                                                  cb_b, nullptr, 1.0f,
                                                  nullptr, ao_f32, nullptr, nullptr);
        }
    }
}
